// NewExpressionAttentionLayer_15908558864390
// MI455X (gfx1250) — matrix operand order repaired and checked by reading, not compiled, not run on hardware
//
#include <hip/hip_runtime.h>
#include <hip/hip_bf16.h>

typedef __attribute__((ext_vector_type(16))) __bf16 v16bf;
typedef __attribute__((ext_vector_type(8)))  __bf16 v8bf;
typedef __attribute__((ext_vector_type(8)))  float  v8f;

#define D_  512
#define H_  8
#define HD_ 64
#define S_  2048
#define B_  4
#define BS_ (B_ * S_)
#define EPS_ 1e-8f
#define VTP 40   // LDS pitch (bf16 elems) for V^T tile: 80B -> 16B aligned, conflict-free
#define PP  40   // LDS pitch for P tile

// ---------- helpers ----------

__device__ __forceinline__ unsigned short f2bf(float f) {
  unsigned int u = __float_as_uint(f);
  u += 0x7FFFu + ((u >> 16) & 1u);   // round-to-nearest-even
  return (unsigned short)(u >> 16);
}

// A-fragment (16x32, 16-bit): lane r = row; slot e holds k = 16g + e.
__device__ __forceinline__ v16bf load_afrag(const unsigned short* base, int g) {
  v8bf lo = *reinterpret_cast<const v8bf*>(base + 16 * g);
  v8bf hi = *reinterpret_cast<const v8bf*>(base + 16 * g + 8);
  v16bf r;
#pragma unroll
  for (int i = 0; i < 8; ++i) { r[i] = lo[i]; r[i + 8] = hi[i]; }
  return r;
}

// B-fragment (32x16, 16-bit): lane r = column n; slot e holds k = 16g + e.
__device__ __forceinline__ v16bf load_bfrag(const unsigned short* p) {
  v8bf lo = *reinterpret_cast<const v8bf*>(p);
  v8bf hi = *reinterpret_cast<const v8bf*>(p + 8);
  v16bf r;
#pragma unroll
  for (int i = 0; i < 8; ++i) { r[i] = lo[i]; r[i + 8] = hi[i]; }
  return r;
}

// ---------- pack kernels ----------

__global__ void pack_inputs_kernel(const float* __restrict__ gene,
                                   const float* __restrict__ expr,
                                   unsigned short* __restrict__ X,
                                   unsigned short* __restrict__ E, int total) {
  int t = blockIdx.x * blockDim.x + threadIdx.x;
  if (t >= total) return;
  int row = t / D_;
  int col = t - row * D_;
  unsigned short gb = f2bf(gene[t]);
  unsigned short eb = f2bf(expr[t]);
  X[(size_t)row * (2 * D_) + col]      = gb;
  X[(size_t)row * (2 * D_) + D_ + col] = eb;
  E[t] = eb;
}

// Wt[n*K + k] = bf16(W[k*N + n])
__global__ void pack_weightT_kernel(const float* __restrict__ W,
                                    unsigned short* __restrict__ Wt, int K, int N) {
  int t = blockIdx.x * blockDim.x + threadIdx.x;
  if (t >= K * N) return;
  int k = t / N;
  int n = t - k * N;
  Wt[(size_t)n * K + k] = f2bf(W[t]);
}

// ---------- bf16 WMMA GEMM: C(MxN) = A(MxK) * Bt^T + bias ----------
// One wave -> 32x64 tile, register double-buffered K pipeline.
__global__ void __launch_bounds__(256)
gemm_bf16_wmma_kernel(const unsigned short* __restrict__ A,
                      const unsigned short* __restrict__ Bt,
                      const float* __restrict__ bias,
                      void* __restrict__ Cout,
                      int M, int N, int K, int c_is_bf16) {
  const int lane = threadIdx.x & 31;
  const int wave = threadIdx.x >> 5;
  const int r = lane & 15;
  const int g = lane >> 4;
  const int n64 = N >> 6;
  const int gw = blockIdx.x * 8 + wave;
  const int tm = gw / n64;
  const int tn = gw - tm * n64;
  if (tm * 32 >= M) return;
  const int m0 = tm * 32, n0 = tn * 64;

  v8f acc0[4] = {v8f{}, v8f{}, v8f{}, v8f{}};
  v8f acc1[4] = {v8f{}, v8f{}, v8f{}, v8f{}};

  const unsigned short* arow0 = A + (size_t)(m0 + r) * K;
  const unsigned short* arow1 = A + (size_t)(m0 + 16 + r) * K;
  const unsigned short* brow[4];
#pragma unroll
  for (int sub = 0; sub < 4; ++sub)
    brow[sub] = Bt + (size_t)(n0 + sub * 16 + r) * K + 16 * g;

  auto load_frags = [&](v16bf& a0, v16bf& a1, v16bf (&bf)[4], int kk) {
    a0 = load_afrag(arow0 + kk, g);
    a1 = load_afrag(arow1 + kk, g);
#pragma unroll
    for (int sub = 0; sub < 4; ++sub) bf[sub] = load_bfrag(brow[sub] + kk);
  };
  auto do_mma = [&](const v16bf& a0, const v16bf& a1, const v16bf (&bf)[4]) {
#pragma unroll
    for (int sub = 0; sub < 4; ++sub) {
      acc0[sub] = __builtin_amdgcn_wmma_f32_16x16x32_bf16(
          false, a0, false, bf[sub], (short)0, acc0[sub], false, false);
      acc1[sub] = __builtin_amdgcn_wmma_f32_16x16x32_bf16(
          false, a1, false, bf[sub], (short)0, acc1[sub], false, false);
    }
  };

  // software pipeline: K/32 is even (K = 512 or 1024)
  v16bf a0A, a1A, a0B, a1B, bfA[4], bfB[4];
  load_frags(a0A, a1A, bfA, 0);
  for (int kk = 0; kk < K; kk += 64) {
    load_frags(a0B, a1B, bfB, kk + 32);
    do_mma(a0A, a1A, bfA);
    if (kk + 64 < K) load_frags(a0A, a1A, bfA, kk + 64);
    do_mma(a0B, a1B, bfB);
  }

#pragma unroll
  for (int sub = 0; sub < 4; ++sub) {
    const int n = n0 + sub * 16 + r;
    const float bv = bias[n];
#pragma unroll
    for (int v = 0; v < 8; ++v) {
      const int row0 = m0 + v + 8 * g;
      const int row1 = row0 + 16;
      const float v0 = acc0[sub][v] + bv;
      const float v1 = acc1[sub][v] + bv;
      if (c_is_bf16) {
        ((unsigned short*)Cout)[(size_t)row0 * N + n] = f2bf(v0);
        ((unsigned short*)Cout)[(size_t)row1 * N + n] = f2bf(v1);
      } else {
        ((float*)Cout)[(size_t)row0 * N + n] = v0;
        ((float*)Cout)[(size_t)row1 * N + n] = v1;
      }
    }
  }
}

// ---------- attention: out = (sum e*m*v) / (sum e*m + EPS * sum e) ----------
__global__ void __launch_bounds__(128)
attn_kernel(const unsigned short* __restrict__ Qb,
            const unsigned short* __restrict__ Kb,
            const unsigned short* __restrict__ Vb,
            const float* __restrict__ Mm,
            unsigned short* __restrict__ Ob) {
  __shared__ unsigned short VT[64 * VTP];      // V^T tile: [dim][key], 32 keys
  __shared__ unsigned short PT[4 * 16 * PP];   // per-wave P tiles: [q][key]

  const int tid = threadIdx.x;
  const int lane = tid & 31;
  const int wave = tid >> 5;
  const int r = lane & 15;
  const int g = lane >> 4;

  const int qblks = S_ / 64;
  const int bid = blockIdx.x;
  const int qb = bid % qblks;
  const int h  = (bid / qblks) % H_;
  const int b  = bid / (qblks * H_);
  const int q0 = qb * 64 + wave * 16;

  const size_t rowQ = (size_t)(b * S_ + q0 + r) * D_ + h * HD_;
  const v16bf aq0 = load_afrag(Qb + rowQ + 0,  g);
  const v16bf aq1 = load_afrag(Qb + rowQ + 32, g);

  float mrow[8], Zr[8], Nr[8];
#pragma unroll
  for (int v = 0; v < 8; ++v) { mrow[v] = -INFINITY; Zr[v] = 0.f; Nr[v] = 0.f; }
  v8f acc[4] = {v8f{}, v8f{}, v8f{}, v8f{}};
  const float scale = 0.125f;  // 1/sqrt(64)
  unsigned short* myP = PT + wave * 16 * PP;

  for (int kt = 0; kt < S_; kt += 32) {
    __syncthreads();  // protect VT against readers of previous iteration
    {   // cooperative transposed staging of V tile (32 keys x 64 dims)
      const int key = tid >> 2;
      const int seg = tid & 3;
      const unsigned short* vsrc =
          Vb + (size_t)(b * S_ + kt + key) * D_ + h * HD_ + seg * 16;
      unsigned short tmp[16];
      *(uint4*)(tmp)     = *(const uint4*)(vsrc);
      *(uint4*)(tmp + 8) = *(const uint4*)(vsrc + 8);
#pragma unroll
      for (int j = 0; j < 16; ++j) VT[(seg * 16 + j) * VTP + key] = tmp[j];
    }
    __syncthreads();

    if (kt + 32 < S_) {  // prefetch next K/V tile rows into caches
      __builtin_prefetch(Kb + (size_t)(b * S_ + kt + 32 + lane) * D_ + h * HD_, 0, 0);
      __builtin_prefetch(Vb + (size_t)(b * S_ + kt + 32 + lane) * D_ + h * HD_, 0, 0);
    }

    // batch all K-fragments, then one WMMA group: S = Q * K^T
    v16bf bk[2][2];
#pragma unroll
    for (int nt = 0; nt < 2; ++nt) {
      const unsigned short* krow =
          Kb + (size_t)(b * S_ + kt + nt * 16 + r) * D_ + h * HD_;
      bk[nt][0] = load_bfrag(krow + 16 * g);
      bk[nt][1] = load_bfrag(krow + 32 + 16 * g);
    }
    v8f sf[2];
#pragma unroll
    for (int nt = 0; nt < 2; ++nt) {
      v8f z = {};
      z = __builtin_amdgcn_wmma_f32_16x16x32_bf16(false, aq0, false, bk[nt][0], (short)0, z, false, false);
      sf[nt] = __builtin_amdgcn_wmma_f32_16x16x32_bf16(false, aq1, false, bk[nt][1], (short)0, z, false, false);
    }

    // scale + mask, online softmax bookkeeping, stage P = e*m
    float sm[2][8], mv[2][8];
#pragma unroll
    for (int nt = 0; nt < 2; ++nt)
#pragma unroll
      for (int v = 0; v < 8; ++v) {
        const size_t mi = (size_t)(b * S_ + q0 + v + 8 * g) * S_ + kt + nt * 16 + r;
        mv[nt][v] = Mm[mi];
        sm[nt][v] = sf[nt][v] * scale * mv[nt][v];
      }

#pragma unroll
    for (int v = 0; v < 8; ++v) {
      float tmax = fmaxf(sm[0][v], sm[1][v]);
#pragma unroll
      for (int off = 1; off < 16; off <<= 1)
        tmax = fmaxf(tmax, __shfl_xor(tmax, off, 32));
      const float nm = fmaxf(mrow[v], tmax);
      const float alpha = __expf(mrow[v] - nm);
      const float e0 = __expf(sm[0][v] - nm);
      const float e1 = __expf(sm[1][v] - nm);
      const float p0 = e0 * mv[0][v];
      const float p1 = e1 * mv[1][v];
      float sz = e0 + e1;
      float sp = p0 + p1;
#pragma unroll
      for (int off = 1; off < 16; off <<= 1) {
        sz += __shfl_xor(sz, off, 32);
        sp += __shfl_xor(sp, off, 32);
      }
      Zr[v] = Zr[v] * alpha + sz;
      Nr[v] = Nr[v] * alpha + sp;
      mrow[v] = nm;
#pragma unroll
      for (int dt = 0; dt < 4; ++dt) acc[dt][v] *= alpha;
      myP[(v + 8 * g) * PP + r]      = f2bf(p0);
      myP[(v + 8 * g) * PP + 16 + r] = f2bf(p1);
    }

    // CDNA5 split-counter wait: DS stores complete before cross-lane DS reads
    asm volatile("s_wait_dscnt 0x0" ::: "memory");

    // batch P/V fragments, then WMMA group: acc += P(16x32) * V(32x64)
    const v16bf ap = load_afrag(myP + r * PP, g);
    v16bf bv[4];
#pragma unroll
    for (int dt = 0; dt < 4; ++dt)
      bv[dt] = load_bfrag(VT + (dt * 16 + r) * VTP + 16 * g);
#pragma unroll
    for (int dt = 0; dt < 4; ++dt)
      acc[dt] = __builtin_amdgcn_wmma_f32_16x16x32_bf16(
          false, ap, false, bv[dt], (short)0, acc[dt], false, false);
  }

  // epilogue: divide by (sum e*m + EPS * sum e), store bf16 head output
#pragma unroll
  for (int v = 0; v < 8; ++v) {
    const float inv = 1.0f / (Nr[v] + EPS_ * Zr[v]);
    const int q = q0 + v + 8 * g;
#pragma unroll
    for (int dt = 0; dt < 4; ++dt)
      Ob[(size_t)(b * S_ + q) * D_ + h * HD_ + dt * 16 + r] = f2bf(acc[dt][v] * inv);
  }
}

// ---------- launch ----------

extern "C" void kernel_launch(void* const* d_in, const int* in_sizes, int n_in,
                              void* d_out, int out_size, void* d_ws, size_t ws_size,
                              hipStream_t stream) {
  (void)in_sizes; (void)n_in; (void)out_size; (void)ws_size;

  const float* gene = (const float*)d_in[0];
  const float* expr = (const float*)d_in[1];
  const float* Mm   = (const float*)d_in[2];
  const float* Wf   = (const float*)d_in[3];
  const float* bfu  = (const float*)d_in[4];
  const float* Wq   = (const float*)d_in[5];
  const float* bq   = (const float*)d_in[6];
  const float* Wk   = (const float*)d_in[7];
  const float* bk   = (const float*)d_in[8];
  const float* Wv   = (const float*)d_in[9];
  const float* bv   = (const float*)d_in[10];
  const float* Wo   = (const float*)d_in[11];
  const float* bo   = (const float*)d_in[12];
  float* out = (float*)d_out;

  char* ws = (char*)d_ws;
  size_t off = 0;
  auto take = [&](size_t elems) -> unsigned short* {
    unsigned short* p = (unsigned short*)(ws + off);
    off += ((elems * 2 + 255) / 256) * 256;
    return p;
  };
  unsigned short* Xbf = take((size_t)BS_ * 2 * D_);  // [gene|expr]
  unsigned short* Ebf = take((size_t)BS_ * D_);
  unsigned short* Wtf = take((size_t)D_ * 2 * D_);   // 512 x 1024
  unsigned short* WtQ = take((size_t)D_ * D_);
  unsigned short* WtK = take((size_t)D_ * D_);
  unsigned short* WtV = take((size_t)D_ * D_);
  unsigned short* WtO = take((size_t)D_ * D_);
  unsigned short* Fbf = take((size_t)BS_ * D_);
  unsigned short* Qbf = take((size_t)BS_ * D_);
  unsigned short* Kbf = take((size_t)BS_ * D_);
  unsigned short* Vbf = take((size_t)BS_ * D_);
  unsigned short* Obf = take((size_t)BS_ * D_);

  const int totE = BS_ * D_;
  pack_inputs_kernel<<<(totE + 255) / 256, 256, 0, stream>>>(gene, expr, Xbf, Ebf, totE);
  pack_weightT_kernel<<<(2 * D_ * D_ + 255) / 256, 256, 0, stream>>>(Wf, Wtf, 2 * D_, D_);
  pack_weightT_kernel<<<(D_ * D_ + 255) / 256, 256, 0, stream>>>(Wq, WtQ, D_, D_);
  pack_weightT_kernel<<<(D_ * D_ + 255) / 256, 256, 0, stream>>>(Wk, WtK, D_, D_);
  pack_weightT_kernel<<<(D_ * D_ + 255) / 256, 256, 0, stream>>>(Wv, WtV, D_, D_);
  pack_weightT_kernel<<<(D_ * D_ + 255) / 256, 256, 0, stream>>>(Wo, WtO, D_, D_);

  const int waves = (BS_ / 32) * (D_ / 64);          // 2048 waves (32x64 tiles)
  const int gblocks = (waves + 7) / 8;               // 256 blocks of 8 waves

  // fused = [gene|expr] @ W_fused + b    (K = 1024)
  gemm_bf16_wmma_kernel<<<gblocks, 256, 0, stream>>>(Xbf, Wtf, bfu, Fbf, BS_, D_, 2 * D_, 1);
  // Q / K / V projections (K = 512)
  gemm_bf16_wmma_kernel<<<gblocks, 256, 0, stream>>>(Fbf, WtQ, bq, Qbf, BS_, D_, D_, 1);
  gemm_bf16_wmma_kernel<<<gblocks, 256, 0, stream>>>(Fbf, WtK, bk, Kbf, BS_, D_, D_, 1);
  gemm_bf16_wmma_kernel<<<gblocks, 256, 0, stream>>>(Ebf, WtV, bv, Vbf, BS_, D_, D_, 1);

  // masked attention with L1 renorm
  attn_kernel<<<B_ * H_ * (S_ / 64), 128, 0, stream>>>(Qbf, Kbf, Vbf, Mm, Obf);

  // final projection, f32 output
  gemm_bf16_wmma_kernel<<<gblocks, 256, 0, stream>>>(Obf, WtO, bo, out, BS_, D_, D_, 0);
}
